// GeneralizedBilinearAttention_81612968558740
// MI455X (gfx1250) — matrix operand order repaired and checked by reading, not compiled, not run on hardware
//
#include <hip/hip_runtime.h>
#include <hip/hip_bf16.h>
#include <math.h>

// Problem constants (match reference)
#define B_  128
#define T_  256
#define N_  128
#define AD  512   // ATTN_DIM
#define VD  512   // VEC_DIM
#define NEG_INF_ -1e30f

typedef __bf16 bf16_t;
typedef __attribute__((ext_vector_type(16))) __bf16 v16bf;
typedef __attribute__((ext_vector_type(8)))  __bf16 v8bf;
typedef __attribute__((ext_vector_type(8)))  float  v8f;
typedef __attribute__((ext_vector_type(4)))  float  v4f;

// ---- helpers -------------------------------------------------------------

static __device__ inline v8bf cvt8(const float* p) {
    // load 8 contiguous f32 (32B, aligned) and convert to bf16
    const v4f a = *reinterpret_cast<const v4f*>(p);
    const v4f b = *reinterpret_cast<const v4f*>(p + 4);
    v8bf r;
    r[0] = (__bf16)a[0]; r[1] = (__bf16)a[1]; r[2] = (__bf16)a[2]; r[3] = (__bf16)a[3];
    r[4] = (__bf16)b[0]; r[5] = (__bf16)b[1]; r[6] = (__bf16)b[2]; r[7] = (__bf16)b[3];
    return r;
}

static __device__ inline v16bf cat16(v8bf lo, v8bf hi) {
    v16bf r;
#pragma unroll
    for (int i = 0; i < 8; ++i) { r[i] = lo[i]; r[8 + i] = hi[i]; }
    return r;
}

static __device__ inline v8f wmma_bf16(v16bf a, v16bf b, v8f c) {
    // D = A(16x32 bf16) * B(32x16 bf16) + C(16x16 f32)
    return __builtin_amdgcn_wmma_f32_16x16x32_bf16(
        /*neg_a=*/false, a, /*neg_b=*/false, b,
        /*c_mod=*/(short)0, c, /*reuse_a=*/false, /*reuse_b=*/false);
}

// A-fragment (16x32 bf16) from an LDS chunk row (padded to CH_PAD elements).
// lane<16: row M=lane%16, K = {16*h..16*h+15}, h = lane>=16.
// CH_PAD=40 -> 80B row stride: the 16 lanes' 16B ds_load_b128 start banks
// (20*r mod 64) form a perfect permutation -> conflict-free.
#define CH_PAD 40
static __device__ inline v16bf load_a_lds(const bf16_t* chrow, int lhalf) {
    return cat16(*reinterpret_cast<const v8bf*>(chrow + 16 * lhalf),
                 *reinterpret_cast<const v8bf*>(chrow + 16 * lhalf + 8));
}

// ---- prep kernels --------------------------------------------------------

// bwT[v][a] = bf16(bi_weight[a][v])   (VD x AD)
__global__ void k_bwT(const float* __restrict__ bw, bf16_t* __restrict__ bwT) {
    int idx = blockIdx.x * 256 + threadIdx.x;      // over VD*AD
    int v = idx / AD, a = idx % AD;
    bwT[idx] = (__bf16)bw[a * VD + v];
}

// aoT[b][a][n] = bf16(attend_over[b][n][a])   (B x AD x N)
__global__ void k_aoT(const float* __restrict__ ao, bf16_t* __restrict__ aoT) {
    long idx = (long)blockIdx.x * 256 + threadIdx.x;   // over B*AD*N
    int  n = (int)(idx % N_);
    long r = idx / N_;
    int  a = (int)(r % AD);
    int  b = (int)(r / AD);
    aoT[idx] = (__bf16)ao[((long)b * N_ + n) * AD + a];
}

// lin_a[b][n] = ao[b,n,:] . a_w ; lin_b[b][t] = inputs[b,t,:] . b_w
__global__ void k_lin(const float* __restrict__ ao, const float* __restrict__ inp,
                      const float* __restrict__ a_w, const float* __restrict__ b_w,
                      float* __restrict__ lin_a, float* __restrict__ lin_b) {
    int idx = blockIdx.x * 256 + threadIdx.x;   // over B*N + B*T rows
    if (idx < B_ * N_) {
        const float* row = ao + (long)idx * AD;
        float s = 0.f;
        for (int i = 0; i < AD; ++i) s += row[i] * a_w[i];
        lin_a[idx] = s;
    } else {
        int j = idx - B_ * N_;
        const float* row = inp + (long)j * VD;
        float s = 0.f;
        for (int i = 0; i < VD; ++i) s += row[i] * b_w[i];
        lin_b[j] = s;
    }
}

// ---- GEMM 1: aM[b][n][v] = sum_a attend_over[b][n][a] * bi_weight[a][v] --
// grid (B, N/16); 256 threads = 8 waves; block tile = 16 (N) x 512 (VD).
// Wave w owns column tiles v0 = (w*4+j)*16, j=0..3. The shared A-fragment
// (16 attend_over rows) is staged in a double-buffered LDS chunk so the
// stage of k-step i+1 overlaps the WMMAs of k-step i (one barrier / step).
__global__ void __launch_bounds__(256)
k_aM(const float* __restrict__ ao, const bf16_t* __restrict__ bwT,
     bf16_t* __restrict__ aM) {
    __shared__ bf16_t chA[2][16][CH_PAD];

    const int b    = blockIdx.x;
    const int n0   = blockIdx.y * 16;
    const int wave = threadIdx.x >> 5;
    const int lane = threadIdx.x & 31;
    const int lhalf = lane >> 4;
    const int lmod  = lane & 15;

    const float* arow_base = ao + ((long)b * N_ + n0) * AD;
    const int sr = threadIdx.x >> 2;   // staging row 0..15 (tid<64 only)
    const int ss = threadIdx.x & 3;    // staging 8-float segment

    const bf16_t* brow[4];
#pragma unroll
    for (int j = 0; j < 4; ++j) {
        const int v0 = (wave * 4 + j) * 16;
        brow[j] = bwT + (long)(v0 + lmod) * AD + 16 * lhalf;
    }

    // prologue: stage chunk k0=0 into buffer 0
    if (threadIdx.x < 64)
        *reinterpret_cast<v8bf*>(&chA[0][sr][ss * 8]) =
            cvt8(arow_base + (long)sr * AD + ss * 8);
    __syncthreads();

    v8f acc[4] = {};
    int p = 0;
    for (int k0 = 0; k0 < AD; k0 += 32) {
        // consume buffer p (ds loads issued before the staging ds_store)
        v16bf afrag = load_a_lds(&chA[p][lmod][0], lhalf);
        // stage next chunk into buffer p^1 (overlaps with WMMAs below)
        if (k0 + 32 < AD && threadIdx.x < 64)
            *reinterpret_cast<v8bf*>(&chA[p ^ 1][sr][ss * 8]) =
                cvt8(arow_base + (long)sr * AD + k0 + 32 + ss * 8);
#pragma unroll
        for (int j = 0; j < 4; ++j) {
            if (k0 + 128 < AD) __builtin_prefetch(brow[j] + k0 + 128, 0, 1);
            v16bf bfrag = *reinterpret_cast<const v16bf*>(brow[j] + k0);
            acc[j] = wmma_bf16(afrag, bfrag, acc[j]);
        }
        __syncthreads();   // next buffer staged + current buffer consumed
        p ^= 1;
    }

    // C layout: VGPR r -> row M = r + 8*lhalf, col N = lmod
#pragma unroll
    for (int j = 0; j < 4; ++j) {
        const int v0 = (wave * 4 + j) * 16;
        bf16_t* orow = aM + ((long)b * N_ + n0 + 8 * lhalf) * VD + v0 + lmod;
#pragma unroll
        for (int r = 0; r < 8; ++r) orow[(long)r * VD] = (__bf16)acc[j][r];
    }
}

// ---- fused: sim -> tanh -> masked softmax -> context ---------------------
// grid (B, T/64); 256 threads = 8 waves. LDS: 64x128 f32 sim tile (padded,
// pad cols double as softmax reduction slots), 64x128 bf16 attn tile,
// double-buffered 64x32 bf16 staged inputs chunk. Total 60.4 KB.
__global__ void __launch_bounds__(256)
k_fused(const float* __restrict__ inp, const unsigned char* __restrict__ mask,
        const bf16_t* __restrict__ aM, const bf16_t* __restrict__ aoT,
        const float* __restrict__ lin_a, const float* __restrict__ lin_b,
        const float* __restrict__ bias, float* __restrict__ out) {
    __shared__ float  simF[64][132];      // 33.8 KB (cols 128..131 = red. slots)
    __shared__ bf16_t attnB[64][128];     // 16 KB
    __shared__ bf16_t chA[2][64][CH_PAD]; // 10 KB double-buffered inputs chunk

    const int b  = blockIdx.x;
    const int t0 = blockIdx.y * 64;
    const int wave = threadIdx.x >> 5;
    const int lane = threadIdx.x & 31;
    const int lhalf = lane >> 4;
    const int lmod  = lane & 15;

    // ---- stage A: sim(64x128) = inputs(64x512) @ aM(128x512)^T ----------
    // wave w owns columns n0w..n0w+15 and all 4 row tiles.
    const int n0w = wave * 16;
    const bf16_t* aMrow = aM + ((long)b * N_ + n0w + lmod) * VD + 16 * lhalf;

    const float* irow_base = inp + ((long)b * T_ + t0) * VD;
    const int sr = threadIdx.x >> 2;   // staging row 0..63
    const int ss = threadIdx.x & 3;    // staging 8-float segment

    // prologue: stage chunk k0=0 into buffer 0
    *reinterpret_cast<v8bf*>(&chA[0][sr][ss * 8]) =
        cvt8(irow_base + (long)sr * VD + ss * 8);
    __syncthreads();

    v8f acc[4] = {};
    int p = 0;
    for (int k0 = 0; k0 < VD; k0 += 32) {
        // consume buffer p: hoist the 4 A-fragments (ds loads before ds_store)
        v16bf afrag[4];
#pragma unroll
        for (int tr = 0; tr < 4; ++tr)
            afrag[tr] = load_a_lds(&chA[p][tr * 16 + lmod][0], lhalf);
        // stage next chunk into buffer p^1 (overlaps with WMMAs below)
        if (k0 + 32 < VD)
            *reinterpret_cast<v8bf*>(&chA[p ^ 1][sr][ss * 8]) =
                cvt8(irow_base + (long)sr * VD + k0 + 32 + ss * 8);
        if (k0 + 128 < VD) __builtin_prefetch(aMrow + k0 + 128, 0, 1);
        v16bf bfrag = *reinterpret_cast<const v16bf*>(aMrow + k0);
#pragma unroll
        for (int tr = 0; tr < 4; ++tr)
            acc[tr] = wmma_bf16(afrag[tr], bfrag, acc[tr]);
        __syncthreads();   // next buffer staged + current buffer consumed
        p ^= 1;
    }

    const float bs = bias[0];
    const float la = lin_a[b * N_ + n0w + lmod];
#pragma unroll
    for (int tr = 0; tr < 4; ++tr) {
#pragma unroll
        for (int r = 0; r < 8; ++r) {
            int trow = tr * 16 + 8 * lhalf + r;
            float v = acc[tr][r] + la + lin_b[b * T_ + t0 + trow] + bs;
            simF[trow][n0w + lmod] = tanhf(v);
        }
    }
    __syncthreads();

    // ---- masked softmax over N: 4 threads per t-row, 32 cols each --------
    {
        const int m = threadIdx.x >> 2;    // row 0..63
        const int q = threadIdx.x & 3;     // quarter 0..3
        const unsigned char* mrow = mask + b * N_ + q * 32;   // jax bool = 1B

        float pmax = NEG_INF_;
#pragma unroll
        for (int i = 0; i < 32; ++i) {
            float v = mrow[i] ? simF[m][q * 32 + i] : NEG_INF_;
            pmax = fmaxf(pmax, v);
        }
        simF[m][128 + q] = pmax;
        __syncthreads();
        float mx = fmaxf(fmaxf(simF[m][128], simF[m][129]),
                         fmaxf(simF[m][130], simF[m][131]));
        float psum = 0.f;
#pragma unroll
        for (int i = 0; i < 32; ++i) {
            int n = q * 32 + i;
            float e = mrow[i] ? __expf(simF[m][n] - mx) : 0.f;
            simF[m][n] = e;                 // each thread owns its 32 columns
            psum += e;
        }
        __syncthreads();                    // all mx reads done before overwrite
        simF[m][128 + q] = psum;
        __syncthreads();
        float sum = simF[m][128] + simF[m][129] + simF[m][130] + simF[m][131];
        float inv = (sum > 0.f) ? (1.f / sum) : 0.f;
#pragma unroll
        for (int i = 0; i < 32; ++i) {
            int n = q * 32 + i;
            attnB[m][n] = (__bf16)(simF[m][n] * inv);
        }
    }
    __syncthreads();

    // ---- stage B: context(64x512) = attn(64x128) @ attend_over(128x512) --
    // B operand from aoT[b][a][n] (bf16, rows contiguous in n = K dim);
    // each wave reads distinct aoT rows, so no cross-wave redundancy.
    // wave w owns columns [w*64, w*64+63] via j = 0..3.
    for (int j = 0; j < 4; ++j) {
        const int a0 = (wave * 4 + j) * 16;
        const bf16_t* brow = aoT + ((long)b * AD + a0 + lmod) * N_ + 16 * lhalf;
        v8f c[4] = {};
        for (int k0 = 0; k0 < N_; k0 += 32) {
            v16bf bfrag = *reinterpret_cast<const v16bf*>(brow + k0);
#pragma unroll
            for (int tr = 0; tr < 4; ++tr) {
                const bf16_t* ar = &attnB[tr * 16 + lmod][0];
                v16bf afrag = cat16(*reinterpret_cast<const v8bf*>(ar + k0 + 16 * lhalf),
                                    *reinterpret_cast<const v8bf*>(ar + k0 + 16 * lhalf + 8));
                c[tr] = wmma_bf16(afrag, bfrag, c[tr]);
            }
        }
#pragma unroll
        for (int tr = 0; tr < 4; ++tr) {
            float* orow = out + ((long)b * T_ + t0 + tr * 16 + 8 * lhalf) * AD + a0 + lmod;
#pragma unroll
            for (int r = 0; r < 8; ++r) orow[(long)r * AD] = c[tr][r];
        }
    }
}

// ---- host launcher -------------------------------------------------------

extern "C" void kernel_launch(void* const* d_in, const int* in_sizes, int n_in,
                              void* d_out, int out_size, void* d_ws, size_t ws_size,
                              hipStream_t stream) {
    const float*         inputs      = (const float*)d_in[0];  // (B,T,VD) f32
    const float*         attend_over = (const float*)d_in[1];  // (B,N,AD) f32
    const unsigned char* attend_mask = (const unsigned char*)d_in[2]; // (B,N) bool
    const float*         bi_weight   = (const float*)d_in[3];  // (AD,VD) f32
    const float*         a_w         = (const float*)d_in[4];  // (AD,)
    const float*         b_w         = (const float*)d_in[5];  // (VD,)
    const float*         bias        = (const float*)d_in[6];  // (1,)
    float*               out         = (float*)d_out;          // (B,T,AD) f32

    // Workspace carve-out (~33 MB total)
    char* ws = (char*)d_ws;
    size_t off = 0;
    bf16_t* bwT   = (bf16_t*)(ws + off); off += (size_t)VD * AD * sizeof(bf16_t);        // 0.5 MB
    bf16_t* aoT   = (bf16_t*)(ws + off); off += (size_t)B_ * AD * N_ * sizeof(bf16_t);   // 16 MB
    bf16_t* aM    = (bf16_t*)(ws + off); off += (size_t)B_ * N_ * VD * sizeof(bf16_t);   // 16 MB
    float*  lin_a = (float*)(ws + off);  off += (size_t)B_ * N_ * sizeof(float);
    float*  lin_b = (float*)(ws + off);  off += (size_t)B_ * T_ * sizeof(float);
    (void)ws_size; (void)in_sizes; (void)n_in; (void)out_size;

    k_bwT<<<(VD * AD) / 256, 256, 0, stream>>>(bi_weight, bwT);
    k_aoT<<<(int)(((long)B_ * AD * N_) / 256), 256, 0, stream>>>(attend_over, aoT);
    k_lin<<<(B_ * N_ + B_ * T_) / 256, 256, 0, stream>>>(attend_over, inputs, a_w, b_w,
                                                         lin_a, lin_b);
    k_aM<<<dim3(B_, N_ / 16), 256, 0, stream>>>(attend_over, bwT, aM);
    k_fused<<<dim3(B_, T_ / 64), 256, 0, stream>>>(inputs, attend_mask, aM, aoT,
                                                   lin_a, lin_b, bias, out);
}
